// ModalityMoERouter_78288663872332
// MI455X (gfx1250) — matrix operand order repaired and checked by reading, not compiled, not run on hardware
//
#include <hip/hip_runtime.h>
#include <math.h>

typedef __attribute__((ext_vector_type(16))) _Float16 v16h;
typedef __attribute__((ext_vector_type(8)))  float    v8f;
typedef __attribute__((ext_vector_type(4)))  int      v4i;

#define BB   8
#define NN   16384
#define DD   256
#define EE   8
#define HIDN 128
#define KSEL 4096
#define MTOT (BB*NN)          // 131072 tokens total

// ---- pre-swizzled weight blob layout (bytes), mirrored 1:1 into LDS ----
// [0      , 73728) : W1 fragments  9 kt x 8 j x 32 lanes x 16 halves
// [73728  , 77824) : W2 fragments  4 kt x 32 lanes x 16 halves
// [77824  , 78336) : b1 (128 f32)
// [78336  , 78368) : b2 (8 f32)
#define W1F_CNT   (9*8*32*16)            // 36864 halves
#define W2F_CNT   (4*32*16)              // 2048 halves
#define BIAS_OFF  ((W1F_CNT + W2F_CNT)*2) // 77824 bytes
#define BLOB_BYTES (BIAS_OFF + (HIDN+EE)*4) // 78368 bytes (=4898 x 16B)
#define BLOB_CHUNKS (BLOB_BYTES/16)      // 4898

// LDS: blob image + per-wave h scratch (A-fragment layout)
#define HSWZ_BYTE 78368
#define HSWZ_H    (HSWZ_BYTE/2)          // 39184 (half index)
#define SMEM_BYTES (HSWZ_BYTE + 8*4*32*16*2)  // +32768 = 111136

// optional CDNA5 async-to-LDS staging path
#if defined(__has_builtin)
#  if __has_builtin(__builtin_amdgcn_global_load_async_to_lds_b128)
#    define USE_ASYNC_LDS 1
#  endif
#  if __has_builtin(__builtin_amdgcn_s_wait_asynccnt)
#    define HAVE_WAIT_ASYNC_BUILTIN 1
#  endif
#endif

using g1_v4i = __attribute__((address_space(1))) v4i;
using l3_v4i = __attribute__((address_space(3))) v4i;

__device__ __forceinline__ unsigned f2key(float f) {
    unsigned u = __float_as_uint(f);
    return (u & 0x80000000u) ? ~u : (u | 0x80000000u);
}

// ---------------- Kernel 0: one-shot weight swizzle into workspace ---------
// B-fragment layout (16-bit, 32xK x 16 tile): lane l holds col n = (l&15);
// k = kt*32 + 16*(v>=4) + 2*(v&3) + (l>>4)*8 + lo  (v = half-pair, lo = low/high half)
__global__ void prep_weights_kernel(const float* __restrict__ W1,
                                    const float* __restrict__ b1,
                                    const float* __restrict__ W2,
                                    const float* __restrict__ b2,
                                    char* __restrict__ blob) {
    int gid = blockIdx.x * 256 + threadIdx.x;
    _Float16* bh = (_Float16*)blob;
    if (gid < W1F_CNT) {
        int i = gid;
        int h16  = i & 15;
        int laneI = (i >> 4) & 31;
        int tile = i >> 9;
        int kt = tile >> 3, j = tile & 7;
        int v = h16 >> 1, lo = h16 & 1;
        int hi = laneI >> 4;
        int n = j*16 + (laneI & 15);
        int k = kt*32 + ((v & 4) ? 16 : 0) + ((v & 3) << 1) + hi*8 + lo;
        float val = (k < DD+3) ? W1[k*HIDN + n] : 0.0f;   // rows 259..287 zero
        bh[i] = (_Float16)val;
    } else if (gid < W1F_CNT + W2F_CNT) {
        int i = gid - W1F_CNT;
        int h16  = i & 15;
        int laneI = (i >> 4) & 31;
        int kt2 = i >> 9;
        int v = h16 >> 1, lo = h16 & 1;
        int hi = laneI >> 4;
        int col = laneI & 15;
        int k = kt2*32 + ((v & 4) ? 16 : 0) + ((v & 3) << 1) + hi*8 + lo;
        float val = (col < EE) ? W2[k*EE + col] : 0.0f;   // cols 8..15 zero
        bh[W1F_CNT + i] = (_Float16)val;
    } else if (gid < W1F_CNT + W2F_CNT + HIDN + EE) {
        int idx = gid - (W1F_CNT + W2F_CNT);
        float val = (idx < HIDN) ? b1[idx] : b2[idx - HIDN];
        ((float*)(blob + BIAS_OFF))[idx] = val;
    }
}

// ---------------- Kernel 1: partial sums of all pairwise distances ----------
__global__ void dist_partial_kernel(const float* __restrict__ xyz,
                                    const float* __restrict__ centers,
                                    float* __restrict__ partials) {
    __shared__ float cen[EE*3];
    __shared__ float red[256];
    int tid = threadIdx.x;
    if (tid < EE*3) cen[tid] = centers[tid];
    __syncthreads();
    int idx = blockIdx.x * 256 + tid;       // token in [0, MTOT)
    float x = xyz[idx*3+0], y = xyz[idx*3+1], z = xyz[idx*3+2];
    float s = 0.f;
    #pragma unroll
    for (int e = 0; e < EE; ++e) {
        float dx = x - cen[e*3+0];
        float dy = y - cen[e*3+1];
        float dz = z - cen[e*3+2];
        s += sqrtf(dx*dx + dy*dy + dz*dz);
    }
    red[tid] = s;
    __syncthreads();
    for (int off = 128; off > 0; off >>= 1) {
        if (tid < off) red[tid] += red[tid + off];
        __syncthreads();
    }
    if (tid == 0) partials[blockIdx.x] = red[0];
}

// ---------------- Kernel 2: finalize mean -> scale = -1/(mean+1e-6) --------
__global__ void dist_finalize_kernel(const float* __restrict__ partials,
                                     float* __restrict__ scale) {
    __shared__ float red[256];
    int tid = threadIdx.x;
    red[tid] = partials[tid] + partials[tid + 256];
    __syncthreads();
    for (int off = 128; off > 0; off >>= 1) {
        if (tid < off) red[tid] += red[tid + off];
        __syncthreads();
    }
    if (tid == 0) {
        float mean = red[0] / (float)((long)MTOT * EE);
        scale[0] = -1.0f / (mean + 1e-6f);
    }
}

// ---------------- Kernel 3: fused gate MLP via WMMA ------------------------
extern __shared__ __align__(32) char smem_raw[];

__global__ __launch_bounds__(256) void moe_gate_kernel(
        const float* __restrict__ tokens,
        const float* __restrict__ xyz,
        const char*  __restrict__ blob,
        const float* __restrict__ centers,
        const float* __restrict__ scale_ptr,
        float* __restrict__ logits_ws,
        float* __restrict__ out) {
    _Float16* sh = (_Float16*)smem_raw;
    const float* sb1 = (const float*)(smem_raw + BIAS_OFF);
    const float* sb2 = sb1 + HIDN;
    int tid = threadIdx.x;

    // ---- stage pre-swizzled weight blob into LDS (flat coalesced copy) ----
#ifdef USE_ASYNC_LDS
    {
        g1_v4i* gp = (g1_v4i*)(uintptr_t)blob;
        for (int i = tid; i < BLOB_CHUNKS; i += 256) {
            l3_v4i* lp = (l3_v4i*)(unsigned int)(uintptr_t)(smem_raw + (size_t)i*16);
            __builtin_amdgcn_global_load_async_to_lds_b128(gp + i, lp, 0, 0);
        }
#  ifdef HAVE_WAIT_ASYNC_BUILTIN
        __builtin_amdgcn_s_wait_asynccnt(0);
#  else
        asm volatile("s_wait_asynccnt 0x0" ::: "memory");
#  endif
    }
#else
    for (int i = tid; i < BLOB_CHUNKS; i += 256) {
        *(float4*)(smem_raw + (size_t)i*16) = *(const float4*)(blob + (size_t)i*16);
    }
#endif
    __syncthreads();

    int lane = tid & 31;
    int wave = tid >> 5;
    int lhi  = lane >> 4;
    int llo  = lane & 15;
    long gRow0 = ((long)blockIdx.x * 8 + wave) * 16;
    const float* rowp = tokens + (gRow0 + llo) * DD;

    v8f zero = {};
    v8f acc[8];
    #pragma unroll
    for (int j = 0; j < 8; ++j) acc[j] = zero;

    // GEMM1 over tokens: 8 k-tiles of 32; A layout:
    // lane l holds row l&15; k_local = 16*(v>=4) + 2*(v&3) + 8*(l>>4) + lo
    #pragma unroll
    for (int kt = 0; kt < 8; ++kt) {
        v16h a;
        #pragma unroll
        for (int v = 0; v < 8; ++v) {
            int k = kt*32 + ((v & 4) ? 16 : 0) + ((v & 3) << 1) + (lhi << 3);
            a[2*v]   = (_Float16)rowp[k];
            a[2*v+1] = (_Float16)rowp[k+1];
        }
        // load all 8 B fragments first so LDS loads overlap the WMMA chain
        v16h bfr[8];
        #pragma unroll
        for (int j = 0; j < 8; ++j)
            bfr[j] = *(const v16h*)&sh[((kt*8 + j)*32 + lane)*16];
        #pragma unroll
        for (int j = 0; j < 8; ++j)
            acc[j] = __builtin_amdgcn_wmma_f32_16x16x32_f16(
                false, a, false, bfr[j], (short)0, acc[j], false, false);
    }
    // 9th k-tile: xyz columns (global k = 256..258), rest zero.
    {
        v16h a;
        #pragma unroll
        for (int i2 = 0; i2 < 16; ++i2) a[i2] = (_Float16)0.0f;
        if (lhi == 0) {
            const float* xp = xyz + (gRow0 + llo) * 3;
            a[0] = (_Float16)xp[0];   // k_local 0 -> k=256
            a[1] = (_Float16)xp[1];   // k_local 1 -> k=257
            a[2] = (_Float16)xp[2];   // k_local 2 -> k=258
        }
        v16h bfr[8];
        #pragma unroll
        for (int j = 0; j < 8; ++j)
            bfr[j] = *(const v16h*)&sh[((8*8 + j)*32 + lane)*16];
        #pragma unroll
        for (int j = 0; j < 8; ++j)
            acc[j] = __builtin_amdgcn_wmma_f32_16x16x32_f16(
                false, a, false, bfr[j], (short)0, acc[j], false, false);
    }

    // bias + exact GELU, write h pre-swizzled (A layout) into per-wave LDS.
    int hbase = HSWZ_H + wave * 2048;
    #pragma unroll
    for (int j = 0; j < 8; ++j) {
        float bbv = sb1[j*16 + llo];
        int n   = j*16 + llo;
        int kt2 = n >> 5;
        int kk  = n & 31;
        int hi2 = (kk >> 3) & 1;
        int v2  = ((kk >> 4) << 2) | ((kk >> 1) & 3);
        int lo2 = kk & 1;
        #pragma unroll
        for (int r = 0; r < 8; ++r) {
            float x = acc[j][r] + bbv;                 // C layout: m=r+8*lhi
            float g = 0.5f * x * (1.0f + erff(x * 0.70710678118654752f));
            int m  = r + (lhi << 3);
            int lp = m | (hi2 << 4);
            sh[hbase + ((kt2*32 + lp)*16 + v2*2 + lo2)] = (_Float16)g;
        }
    }
    __syncthreads();

    // GEMM2: logits(16x16, cols 0..7 valid) = h(16x128) @ W2pad(128x16)
    v8f acc2 = zero;
    {
        v16h ah[4], bw[4];
        #pragma unroll
        for (int kt2 = 0; kt2 < 4; ++kt2) {
            ah[kt2] = *(const v16h*)&sh[hbase + (kt2*32 + lane)*16];
            bw[kt2] = *(const v16h*)&sh[W1F_CNT + (kt2*32 + lane)*16];
        }
        #pragma unroll
        for (int kt2 = 0; kt2 < 4; ++kt2)
            acc2 = __builtin_amdgcn_wmma_f32_16x16x32_f16(
                false, ah[kt2], false, bw[kt2], (short)0, acc2, false, false);
    }

    // Epilogue: + b2 + spatial affinity; write logits (B,E,N) and floor out.
    float scl = scale_ptr[0];
    int e = llo;
    if (e < EE) {
        float cx = centers[e*3+0], cy = centers[e*3+1], cz = centers[e*3+2];
        float b2v = sb2[e];
        long bIdx = gRow0 >> 14;                 // / NN
        #pragma unroll
        for (int r = 0; r < 8; ++r) {
            long grow = gRow0 + r + (lhi << 3);
            float tx = xyz[grow*3+0], ty = xyz[grow*3+1], tz = xyz[grow*3+2];
            float dx = tx-cx, dy = ty-cy, dz = tz-cz;
            float dist = sqrtf(dx*dx + dy*dy + dz*dz);
            float lg = acc2[r] + b2v + dist * scl;
            long tok = grow - bIdx * NN;
            logits_ws[(bIdx*EE + e)*NN + tok] = lg;
            out[grow*EE + e] = 0.0375f;          // routing floor alpha/E
        }
    }
}

// ---------------- Kernel 4: expert-choice top-k select + scatter -----------
__global__ void topk_scatter_kernel(const float* __restrict__ logits_ws,
                                    float* __restrict__ out) {
    __shared__ unsigned hist[256];
    __shared__ unsigned sh_chosen, sh_r, eqcount;
    __shared__ int eqlist[256];
    int tid = threadIdx.x;
    int p = blockIdx.x;                  // b*EE + e
    int b = p >> 3, e = p & 7;
    const float* row = logits_ws + (long)p * NN;
    float* outb = out + (long)b * NN * EE + e;

    // radix select: find 4096th-largest key, 8 bits per pass from MSB
    unsigned prefix = 0, pmask = 0, r = KSEL;
    for (int shift = 24; shift >= 0; shift -= 8) {
        hist[tid] = 0;
        __syncthreads();
        for (int n = tid; n < NN; n += 256) {
            unsigned key = f2key(row[n]);
            if ((key & pmask) == prefix)
                atomicAdd(&hist[(key >> shift) & 255], 1u);
        }
        __syncthreads();
        if (tid == 0) {
            unsigned cum = 0, chosen = 0, rr = r;
            for (int bb = 255; bb >= 0; --bb) {
                unsigned c = hist[bb];
                if (cum + c >= r) { chosen = (unsigned)bb; rr = r - cum; break; }
                cum += c;
            }
            sh_chosen = chosen; sh_r = rr; eqcount = 0;
        }
        __syncthreads();
        prefix |= sh_chosen << shift;
        pmask  |= (0xFFu << shift);
        r = sh_r;
    }
    unsigned Kth = prefix;       // exact threshold key
    unsigned sel_eq = r;         // how many equal-to-threshold to take (>=1)

    for (int n = tid; n < NN; n += 256) {
        float v = row[n];
        unsigned key = f2key(v);
        if (key > Kth) {
            outb[(long)n * EE] = 0.7f * (1.0f / (1.0f + expf(-v))) + 0.0375f;
        } else if (key == Kth) {
            unsigned i = atomicAdd(&eqcount, 1u);
            if (i < 256u) eqlist[i] = n;
        }
    }
    __syncthreads();
    if (tid == 0) {
        int cnt = (int)(eqcount < 256u ? eqcount : 256u);
        for (int i = 1; i < cnt; ++i) {           // sort by index: deterministic
            int key = eqlist[i]; int jj = i - 1;
            while (jj >= 0 && eqlist[jj] > key) { eqlist[jj+1] = eqlist[jj]; --jj; }
            eqlist[jj+1] = key;
        }
        int take = (int)sel_eq; if (take > cnt) take = cnt;
        for (int i = 0; i < take; ++i) {
            int n = eqlist[i];
            float v = row[n];
            outb[(long)n * EE] = 0.7f * (1.0f / (1.0f + expf(-v))) + 0.0375f;
        }
    }
}

// ---------------- host entry ----------------
extern "C" void kernel_launch(void* const* d_in, const int* in_sizes, int n_in,
                              void* d_out, int out_size, void* d_ws, size_t ws_size,
                              hipStream_t stream) {
    const float* tokens  = (const float*)d_in[0];
    const float* xyz     = (const float*)d_in[1];
    const float* W1      = (const float*)d_in[2];
    const float* b1      = (const float*)d_in[3];
    const float* W2      = (const float*)d_in[4];
    const float* b2      = (const float*)d_in[5];
    const float* centers = (const float*)d_in[6];
    // d_in[7] (t) unused by the reference computation
    float* out = (float*)d_out;

    char*  ws        = (char*)d_ws;
    float* logits_ws = (float*)ws;                                  // B*E*N f32 = 4 MiB
    float* partials  = (float*)(ws + (size_t)BB * EE * NN * 4);     // 512 f32
    float* scale     = partials + 512;                              // 1 f32
    char*  blob      = ws + (size_t)BB * EE * NN * 4 + 4096;        // 78368 B

    prep_weights_kernel<<<(W1F_CNT + W2F_CNT + HIDN + EE + 255)/256, 256, 0, stream>>>(
        W1, b1, W2, b2, blob);
    dist_partial_kernel<<<MTOT/256, 256, 0, stream>>>(xyz, centers, partials);
    dist_finalize_kernel<<<1, 256, 0, stream>>>(partials, scale);
    moe_gate_kernel<<<MTOT/128, 256, SMEM_BYTES, stream>>>(
        tokens, xyz, blob, centers, scale, logits_ws, out);
    topk_scatter_kernel<<<BB*EE, 256, 0, stream>>>(logits_ws, out);
}
